// XCABlock_16114717295154
// MI455X (gfx1250) — compile-verified, matrix operand order checked by reading, not run on hardware
//
#include <hip/hip_runtime.h>
#include <hip/hip_bf16.h>
#include <math.h>

typedef __attribute__((ext_vector_type(16))) _Float16 v16h;
typedef __attribute__((ext_vector_type(8)))  _Float16 v8h;
typedef __attribute__((ext_vector_type(8)))  float    v8f;

// Problem dims (fixed by setup_inputs)
#define BB    16
#define HH    56
#define WW    56
#define CC    256
#define NN    (HH*WW)          // 3136
#define MM    (BB*NN)          // 50176
#define HEADS 32
#define DH    8
#define CM    1024
#define BCHW  (BB*CC*NN)       // 12,845,056

__device__ __forceinline__ float gelu_exact(float x) {
    return 0.5f * x * (1.0f + erff(x * 0.70710678118654752440f));
}
#define BNSCALE 0.99999500003749981f   // 1/sqrt(1+1e-5)

// ---------------------------------------------------------------------------
// Weight transpose + f32->f16:  Wt[n*K + k] = (f16) W[k*N + n]
__global__ void __launch_bounds__(256) pack_wt(const float* __restrict__ W,
                                               _Float16* __restrict__ Wt,
                                               int K, int Nn) {
    int idx = blockIdx.x * 256 + threadIdx.x;
    if (idx < K * Nn) {
        int k = idx / Nn, n = idx % Nn;
        Wt[(size_t)n * K + k] = (_Float16)W[idx];
    }
}

// ---------------------------------------------------------------------------
// LayerNorm over C=256, one block per row, output f16 [M,256]
__global__ void __launch_bounds__(256) ln256_to_f16(const float* __restrict__ src,
                                                    const float* __restrict__ w,
                                                    const float* __restrict__ b,
                                                    _Float16* __restrict__ dst) {
    __shared__ float s1[256], s2[256];
    int row = blockIdx.x, c = threadIdx.x;
    float v = src[(size_t)row * CC + c];
    s1[c] = v; s2[c] = v * v; __syncthreads();
    for (int off = 128; off > 0; off >>= 1) {
        if (c < off) { s1[c] += s1[c + off]; s2[c] += s2[c + off]; }
        __syncthreads();
    }
    float mu  = s1[0] * (1.0f / CC);
    float var = s2[0] * (1.0f / CC) - mu * mu;
    float y = (v - mu) * rsqrtf(var + 1e-5f) * w[c] + b[c];
    dst[(size_t)row * CC + c] = (_Float16)y;
}

// LayerNorm over C=256, one block per row, output f32 NCHW (Token2Patch)
__global__ void __launch_bounds__(256) ln256_to_nchw(const float* __restrict__ src,
                                                     const float* __restrict__ w,
                                                     const float* __restrict__ b,
                                                     float* __restrict__ dst) {
    __shared__ float s1[256], s2[256];
    int row = blockIdx.x, c = threadIdx.x;
    float v = src[(size_t)row * CC + c];
    s1[c] = v; s2[c] = v * v; __syncthreads();
    for (int off = 128; off > 0; off >>= 1) {
        if (c < off) { s1[c] += s1[c + off]; s2[c] += s2[c + off]; }
        __syncthreads();
    }
    float mu  = s1[0] * (1.0f / CC);
    float var = s2[0] * (1.0f / CC) - mu * mu;
    float y = (v - mu) * rsqrtf(var + 1e-5f) * w[c] + b[c];
    int bb = row / NN, n = row % NN;
    dst[((size_t)(bb * CC + c)) * NN + n] = y;
}

// GELU then LayerNorm over 1024, one block per row (4 elems/thread), output f16
__global__ void __launch_bounds__(256) gelu_ln1024_f16(const float* __restrict__ src,
                                                       const float* __restrict__ w,
                                                       const float* __restrict__ b,
                                                       _Float16* __restrict__ dst) {
    __shared__ float s1[256], s2[256];
    int row = blockIdx.x, t = threadIdx.x;
    float v[4]; float ls = 0.f, lq = 0.f;
#pragma unroll
    for (int i = 0; i < 4; i++) {
        float x = src[(size_t)row * CM + t + 256 * i];
        float g = gelu_exact(x);
        v[i] = g; ls += g; lq += g * g;
    }
    s1[t] = ls; s2[t] = lq; __syncthreads();
    for (int off = 128; off > 0; off >>= 1) {
        if (t < off) { s1[t] += s1[t + off]; s2[t] += s2[t + off]; }
        __syncthreads();
    }
    float mu  = s1[0] * (1.0f / CM);
    float var = s2[0] * (1.0f / CM) - mu * mu;
    float rs  = rsqrtf(var + 1e-5f);
#pragma unroll
    for (int i = 0; i < 4; i++) {
        int cc = t + 256 * i;
        dst[(size_t)row * CM + cc] = (_Float16)((v[i] - mu) * rs * w[cc] + b[cc]);
    }
}

// ---------------------------------------------------------------------------
// Fragment loader: two contiguous 16B loads -> WMMA 16-bit operand layout
__device__ __forceinline__ v16h load_frag(const _Float16* __restrict__ p) {
    v8h lo = *(const v8h*)(p);
    v8h hi = *(const v8h*)(p + 16);
    v16h f;
#pragma unroll
    for (int i = 0; i < 8; i++) { f[i] = lo[i]; f[i + 8] = hi[i]; }
    return f;
}

// WMMA GEMM: C[M,N] = A[M,K](f16) * Wt[N,K](f16, pre-transposed) + bias[N]
// One wave computes a 16x64 tile. Software-pipelined K-loop: fragments for
// k-step s+1 are loaded while the 4 WMMAs for step s execute.
__global__ void __launch_bounds__(32) gemm_wmma_f16(const _Float16* __restrict__ A,
                                                    const _Float16* __restrict__ Wt,
                                                    const float* __restrict__ bias,
                                                    float* __restrict__ Cout,
                                                    int K, int Nout) {
    int lane  = threadIdx.x;
    int mbase = blockIdx.x * 16;
    int nbase = blockIdx.y * 64;
    int hsel  = lane >> 4;       // 0: K{0..7,16..23}, 1: K{8..15,24..31}
    int l16   = lane & 15;

    const _Float16* Aptr = A + (size_t)(mbase + l16) * K + hsel * 8;
    const _Float16* Bp0  = Wt + (size_t)(nbase +  0 + l16) * K + hsel * 8;
    const _Float16* Bp1  = Wt + (size_t)(nbase + 16 + l16) * K + hsel * 8;
    const _Float16* Bp2  = Wt + (size_t)(nbase + 32 + l16) * K + hsel * 8;
    const _Float16* Bp3  = Wt + (size_t)(nbase + 48 + l16) * K + hsel * 8;

    v8f acc0 = {}, acc1 = {}, acc2 = {}, acc3 = {};

    // prologue: fragments for k-step 0
    v16h a_c  = load_frag(Aptr);
    v16h b0_c = load_frag(Bp0);
    v16h b1_c = load_frag(Bp1);
    v16h b2_c = load_frag(Bp2);
    v16h b3_c = load_frag(Bp3);

    int nk = K >> 5;
    for (int ks = 1; ks < nk; ks++) {
        int k0 = ks * 32;
        // prefetch next k-step while current WMMAs run
        v16h a_n  = load_frag(Aptr + k0);
        v16h b0_n = load_frag(Bp0 + k0);
        v16h b1_n = load_frag(Bp1 + k0);
        v16h b2_n = load_frag(Bp2 + k0);
        v16h b3_n = load_frag(Bp3 + k0);

        acc0 = __builtin_amdgcn_wmma_f32_16x16x32_f16(false, a_c, false, b0_c, (short)0, acc0, false, false);
        acc1 = __builtin_amdgcn_wmma_f32_16x16x32_f16(false, a_c, false, b1_c, (short)0, acc1, false, false);
        acc2 = __builtin_amdgcn_wmma_f32_16x16x32_f16(false, a_c, false, b2_c, (short)0, acc2, false, false);
        acc3 = __builtin_amdgcn_wmma_f32_16x16x32_f16(false, a_c, false, b3_c, (short)0, acc3, false, false);

        a_c = a_n; b0_c = b0_n; b1_c = b1_n; b2_c = b2_n; b3_c = b3_n;
    }
    // epilogue
    acc0 = __builtin_amdgcn_wmma_f32_16x16x32_f16(false, a_c, false, b0_c, (short)0, acc0, false, false);
    acc1 = __builtin_amdgcn_wmma_f32_16x16x32_f16(false, a_c, false, b1_c, (short)0, acc1, false, false);
    acc2 = __builtin_amdgcn_wmma_f32_16x16x32_f16(false, a_c, false, b2_c, (short)0, acc2, false, false);
    acc3 = __builtin_amdgcn_wmma_f32_16x16x32_f16(false, a_c, false, b3_c, (short)0, acc3, false, false);

    v8f accs[4] = {acc0, acc1, acc2, acc3};
#pragma unroll
    for (int t = 0; t < 4; t++) {
        int col = nbase + t * 16 + l16;
        float bv = bias[col];
#pragma unroll
        for (int r = 0; r < 8; r++) {
            int row = mbase + r + hsel * 8;
            Cout[(size_t)row * Nout + col] = accs[t][r] + bv;
        }
    }
}

// ---------------------------------------------------------------------------
// XCA attention: per (b,h) block. qkv: f32 [B*N, 768].
// Writes attn@v into the "exact source reshape" flat layout as f16:
//   outA[b*N*C + (h*8+i)*N + n]
__global__ void __launch_bounds__(256) xca_attn(const float* __restrict__ qkv,
                                                const float* __restrict__ temperature,
                                                _Float16* __restrict__ outA) {
    __shared__ float red[8 * 80];
    __shared__ float fin[80];
    __shared__ float attn[64];

    int b = blockIdx.x, h = blockIdx.y, tid = threadIdx.x;
    const float* base = qkv + (size_t)b * NN * (3 * CC);

    float qq[8] = {}, kk[8] = {}, qk[64] = {};
    for (int n = tid; n < NN; n += 256) {
        const float* r = base + (size_t)n * (3 * CC);
        float q[8], k[8];
#pragma unroll
        for (int i = 0; i < 8; i++) { q[i] = r[h * 8 + i]; k[i] = r[CC + h * 8 + i]; }
#pragma unroll
        for (int i = 0; i < 8; i++) {
            qq[i] += q[i] * q[i];
            kk[i] += k[i] * k[i];
#pragma unroll
            for (int j = 0; j < 8; j++) qk[i * 8 + j] += q[i] * k[j];
        }
    }
    // wave32 reduce each of the 80 partials, then cross-wave reduce in LDS
    int wave = tid >> 5;
#pragma unroll
    for (int rr = 0; rr < 80; rr++) {
        float v = (rr < 8) ? qq[rr] : (rr < 16) ? kk[rr - 8] : qk[rr - 16];
        for (int m = 16; m > 0; m >>= 1) v += __shfl_xor(v, m, 32);
        if ((tid & 31) == 0) red[wave * 80 + rr] = v;
    }
    __syncthreads();
    if (tid < 80) {
        float s = 0.f;
#pragma unroll
        for (int w = 0; w < 8; w++) s += red[w * 80 + tid];
        fin[tid] = s;
    }
    __syncthreads();
    if (tid < 8) {
        int i = tid;
        float nq = fmaxf(sqrtf(fin[i]), 1e-12f);
        float temp = temperature[h];
        float e[8]; float mx = -1e30f;
#pragma unroll
        for (int j = 0; j < 8; j++) {
            float nk = fmaxf(sqrtf(fin[8 + j]), 1e-12f);
            float a = temp * fin[16 + i * 8 + j] / (nq * nk);
            e[j] = a; mx = fmaxf(mx, a);
        }
        float s = 0.f;
#pragma unroll
        for (int j = 0; j < 8; j++) { e[j] = expf(e[j] - mx); s += e[j]; }
        float inv = 1.0f / s;
#pragma unroll
        for (int j = 0; j < 8; j++) attn[i * 8 + j] = e[j] * inv;
    }
    __syncthreads();
    _Float16* ob = outA + (size_t)b * NN * CC;
    for (int n = tid; n < NN; n += 256) {
        const float* r = base + (size_t)n * (3 * CC) + 2 * CC + h * 8;
        float v[8];
#pragma unroll
        for (int j = 0; j < 8; j++) v[j] = r[j];
#pragma unroll
        for (int i = 0; i < 8; i++) {
            float s = 0.f;
#pragma unroll
            for (int j = 0; j < 8; j++) s += attn[i * 8 + j] * v[j];
            ob[(size_t)(h * 8 + i) * NN + n] = (_Float16)s;
        }
    }
}

// ---------------------------------------------------------------------------
// Residual: xbuf = xin + gamma[c]*add    (all f32 [M,256])
__global__ void __launch_bounds__(256) resid_add(const float* __restrict__ xin,
                                                 const float* __restrict__ gamma,
                                                 const float* __restrict__ add,
                                                 float* __restrict__ xbuf) {
    int idx = blockIdx.x * 256 + threadIdx.x;
    if (idx < MM * CC) {
        int c = idx & (CC - 1);
        xbuf[idx] = xin[idx] + gamma[c] * add[idx];
    }
}

// Final: out = xbuf + gamma[c]*out (out pre-filled by fc2 GEMM)
__global__ void __launch_bounds__(256) resid_final(const float* __restrict__ xbuf,
                                                   const float* __restrict__ gamma,
                                                   float* __restrict__ out) {
    int idx = blockIdx.x * 256 + threadIdx.x;
    if (idx < MM * CC) {
        int c = idx & (CC - 1);
        out[idx] = xbuf[idx] + gamma[c] * out[idx];
    }
}

// ---------------------------------------------------------------------------
// Depthwise 3x3, padding 1, NCHW. Variant 1: +bias -> GELU -> BN-eval scale.
__global__ void __launch_bounds__(256) dw3x3_gelu(const float* __restrict__ in,
                                                  const float* __restrict__ w,
                                                  const float* __restrict__ bias,
                                                  float* __restrict__ out) {
    int idx = blockIdx.x * 256 + threadIdx.x;
    if (idx >= BCHW) return;
    int xw = idx % WW;
    int y  = (idx / WW) % HH;
    int bc = idx / NN;
    int c  = bc % CC;
    const float* wp = w + c * 9;
    const float* ip = in + (size_t)bc * NN;
    float acc = bias[c];
#pragma unroll
    for (int dy = 0; dy < 3; dy++) {
        int yy = y + dy - 1;
        if (yy < 0 || yy >= HH) continue;
#pragma unroll
        for (int dx = 0; dx < 3; dx++) {
            int xx = xw + dx - 1;
            if (xx < 0 || xx >= WW) continue;
            acc += ip[yy * WW + xx] * wp[dy * 3 + dx];
        }
    }
    out[idx] = gelu_exact(acc) * BNSCALE;
}

// Variant 2: +bias, then residual scatter back to token layout:
// xbuf[b, n, c] += gamma2[c]*acc
__global__ void __launch_bounds__(256) dw3x3_resid(const float* __restrict__ in,
                                                   const float* __restrict__ w,
                                                   const float* __restrict__ bias,
                                                   const float* __restrict__ gamma,
                                                   float* __restrict__ xbuf) {
    int idx = blockIdx.x * 256 + threadIdx.x;
    if (idx >= BCHW) return;
    int xw = idx % WW;
    int y  = (idx / WW) % HH;
    int bc = idx / NN;
    int c  = bc % CC;
    int bb = bc / CC;
    const float* wp = w + c * 9;
    const float* ip = in + (size_t)bc * NN;
    float acc = bias[c];
#pragma unroll
    for (int dy = 0; dy < 3; dy++) {
        int yy = y + dy - 1;
        if (yy < 0 || yy >= HH) continue;
#pragma unroll
        for (int dx = 0; dx < 3; dx++) {
            int xx = xw + dx - 1;
            if (xx < 0 || xx >= WW) continue;
            acc += ip[yy * WW + xx] * wp[dy * 3 + dx];
        }
    }
    int n = y * WW + xw;
    size_t o = ((size_t)bb * NN + n) * CC + c;
    xbuf[o] += gamma[c] * acc;
}

// ---------------------------------------------------------------------------
extern "C" void kernel_launch(void* const* d_in, const int* in_sizes, int n_in,
                              void* d_out, int out_size, void* d_ws, size_t ws_size,
                              hipStream_t stream) {
    const float* x       = (const float*)d_in[0];
    const float* ln1_w   = (const float*)d_in[3];
    const float* ln1_b   = (const float*)d_in[4];
    const float* Wqkv    = (const float*)d_in[5];
    const float* bqkv    = (const float*)d_in[6];
    const float* temp    = (const float*)d_in[7];
    const float* Wproj   = (const float*)d_in[8];
    const float* bproj   = (const float*)d_in[9];
    const float* ln2_w   = (const float*)d_in[10];
    const float* ln2_b   = (const float*)d_in[11];
    const float* dw1_w   = (const float*)d_in[12];
    const float* dw1_b   = (const float*)d_in[13];
    const float* dw2_w   = (const float*)d_in[14];
    const float* dw2_b   = (const float*)d_in[15];
    const float* ln3_w   = (const float*)d_in[16];
    const float* ln3_b   = (const float*)d_in[17];
    const float* fc1_w   = (const float*)d_in[18];
    const float* fc1_b   = (const float*)d_in[19];
    const float* mlpln_w = (const float*)d_in[20];
    const float* mlpln_b = (const float*)d_in[21];
    const float* fc2_w   = (const float*)d_in[22];
    const float* fc2_b   = (const float*)d_in[23];
    const float* gamma1  = (const float*)d_in[24];
    const float* gamma2  = (const float*)d_in[25];
    const float* gamma3  = (const float*)d_in[26];
    float* out = (float*)d_out;

    // Workspace layout
    char* ws = (char*)d_ws;
    float*    xbuf = (float*)ws;                                     // [M,256] f32
    _Float16* a16  = (_Float16*)(ws + (size_t)MM * CC * 4);          // [M,1024] f16
    float*    g32  = (float*)(ws + (size_t)MM * CC * 4
                                 + (size_t)MM * CM * 2);             // [M,1024] f32
    _Float16* w16  = (_Float16*)(ws + (size_t)MM * CC * 4
                                    + (size_t)MM * CM * 2
                                    + (size_t)MM * CM * 4);          // packed weights
    _Float16* wt_qkv  = w16;
    _Float16* wt_proj = wt_qkv  + (size_t)CC * 3 * CC;   // 196608
    _Float16* wt_fc1  = wt_proj + (size_t)CC * CC;       // +65536
    _Float16* wt_fc2  = wt_fc1  + (size_t)CC * CM;       // +262144
    float* t2 = g32;                       // [B,C,H,W]
    float* t3 = g32 + (size_t)MM * CC;     // [B,C,H,W]

    const int EW = 256;
    // 0) pack weights to f16 column-major
    pack_wt<<<(CC * 3 * CC + EW - 1) / EW, EW, 0, stream>>>(Wqkv,  wt_qkv,  CC, 3 * CC);
    pack_wt<<<(CC * CC     + EW - 1) / EW, EW, 0, stream>>>(Wproj, wt_proj, CC, CC);
    pack_wt<<<(CC * CM     + EW - 1) / EW, EW, 0, stream>>>(fc1_w, wt_fc1,  CC, CM);
    pack_wt<<<(CM * CC     + EW - 1) / EW, EW, 0, stream>>>(fc2_w, wt_fc2,  CM, CC);

    // ---- branch 1: XCA ----
    ln256_to_f16<<<MM, 256, 0, stream>>>(x, ln1_w, ln1_b, a16);
    gemm_wmma_f16<<<dim3(MM / 16, (3 * CC) / 64), 32, 0, stream>>>(a16, wt_qkv, bqkv, g32, CC, 3 * CC);
    xca_attn<<<dim3(BB, HEADS), 256, 0, stream>>>(g32, temp, a16);
    gemm_wmma_f16<<<dim3(MM / 16, CC / 64), 32, 0, stream>>>(a16, wt_proj, bproj, g32, CC, CC);
    resid_add<<<(MM * CC + EW - 1) / EW, EW, 0, stream>>>(x, gamma1, g32, xbuf);

    // ---- branch 2: LPI ----
    ln256_to_nchw<<<MM, 256, 0, stream>>>(xbuf, ln2_w, ln2_b, t2);
    dw3x3_gelu<<<(BCHW + EW - 1) / EW, EW, 0, stream>>>(t2, dw1_w, dw1_b, t3);
    dw3x3_resid<<<(BCHW + EW - 1) / EW, EW, 0, stream>>>(t3, dw2_w, dw2_b, gamma2, xbuf);

    // ---- branch 3: MLP ----
    ln256_to_f16<<<MM, 256, 0, stream>>>(xbuf, ln3_w, ln3_b, a16);
    gemm_wmma_f16<<<dim3(MM / 16, CM / 64), 32, 0, stream>>>(a16, wt_fc1, fc1_b, g32, CC, CM);
    gelu_ln1024_f16<<<MM, 256, 0, stream>>>(g32, mlpln_w, mlpln_b, a16);
    gemm_wmma_f16<<<dim3(MM / 16, CC / 64), 32, 0, stream>>>(a16, wt_fc2, fc2_b, out, CM, CC);
    resid_final<<<(MM * CC + EW - 1) / EW, EW, 0, stream>>>(xbuf, gamma3, out);
}
